// _GraphAttentionLayer_56049323213422
// MI455X (gfx1250) — matrix operand order repaired and checked by reading, not compiled, not run on hardware
//
#include <hip/hip_runtime.h>
#include <hip/hip_bf16.h>

// ---------------------------------------------------------------------------
// GraphAttentionLayer for MI455X (gfx1250, wave32, WMMA).
// N=50000 nodes, E=800000 edges, IN=512, H=8, D=32, EDGE_DIM=32.
// ---------------------------------------------------------------------------

typedef __attribute__((ext_vector_type(16))) __bf16        v16bf;
typedef __attribute__((ext_vector_type(8)))  unsigned int  v8u;    // 32B raw
typedef __attribute__((ext_vector_type(8)))  float         v8f;

// ---------------------------------------------------------------------------
// Kernel 0: one-time W (512x256 f32) -> bf16 copy. 256 KB, L2-resident after.
// ---------------------------------------------------------------------------
__global__ void gat_cvt_w_bf16(const float* __restrict__ W,
                               __bf16* __restrict__ Wb, int total) {
  const int i = blockIdx.x * blockDim.x + threadIdx.x;
  if (i < total) Wb[i] = (__bf16)W[i];
}

// ---------------------------------------------------------------------------
// Kernel 1: Wh = X @ W  (M x 512)*(512 x 256) via v_wmma_f32_16x16x32_bf16.
// Block = 256 threads = 8 waves arranged as 4 M-strips x 2 N-halves over a
// 64x256 block tile. Each wave: 16 rows x 128 cols = 8 f32 accumulators
// (64 VGPRs -> no spills), one A fragment amortized over 8 WMMAs. The two
// waves sharing a row strip reuse A lines via the WGP cache, so X is read
// ~once from HBM; W is bf16 and L2-resident.
// ---------------------------------------------------------------------------
__global__ void __launch_bounds__(256)
gat_gemm_wmma_bf16(const float* __restrict__ X,
                   const __bf16* __restrict__ Wb,
                   float* __restrict__ Wh, int M) {
  constexpr int K = 512, N = 256;
  const int wave = threadIdx.x >> 5;
  const int lane = threadIdx.x & 31;
  const int lo   = lane & 15;
  const int hi   = lane >> 4;                    // 0 or 1

  const int m0 = blockIdx.x * 64 + (wave >> 1) * 16;   // 4 M-strips
  const int n0 = (wave & 1) * 128;                     // 2 N-halves

  // A-fragment row (clamped for ragged M edge; stores are predicated).
  int arow = m0 + lo;
  if (arow >= M) arow = M - 1;
  const float* __restrict__ aptr = X + (size_t)arow * K;

  v8f acc[8];
#pragma unroll
  for (int t = 0; t < 8; ++t) acc[t] = (v8f){};

  for (int k0 = 0; k0 < K; k0 += 32) {
    // ---- A fragment: 16x32 bf16 (ISA 7.12.2): halves 0..7 = K[base..base+7],
    //      halves 8..15 = K[base+16..base+23], base = hi*8.
    const int ka = k0 + hi * 8;
    float4 a0 = *(const float4*)(aptr + ka);
    float4 a1 = *(const float4*)(aptr + ka + 4);
    float4 a2 = *(const float4*)(aptr + ka + 16);
    float4 a3 = *(const float4*)(aptr + ka + 20);
    v16bf av;
    av[0]=(__bf16)a0.x;  av[1]=(__bf16)a0.y;  av[2]=(__bf16)a0.z;  av[3]=(__bf16)a0.w;
    av[4]=(__bf16)a1.x;  av[5]=(__bf16)a1.y;  av[6]=(__bf16)a1.z;  av[7]=(__bf16)a1.w;
    av[8]=(__bf16)a2.x;  av[9]=(__bf16)a2.y;  av[10]=(__bf16)a2.z; av[11]=(__bf16)a2.w;
    av[12]=(__bf16)a3.x; av[13]=(__bf16)a3.y; av[14]=(__bf16)a3.z; av[15]=(__bf16)a3.w;

    // ---- B fragments: lane holds column n = n0 + t*16 + lo; elements 0..7 =
    //      K[base..base+7], 8..15 = K[base+16..base+23], base = hi*8 (matches A).
    const __bf16* __restrict__ brow = Wb + (size_t)(k0 + hi * 8) * N + n0 + lo;
#pragma unroll
    for (int t = 0; t < 8; ++t) {
      v16bf bv;
#pragma unroll
      for (int s = 0; s < 8; ++s) {
        bv[s]     = brow[(size_t)s * N + t * 16];
        bv[8 + s] = brow[(size_t)(16 + s) * N + t * 16];
      }
      acc[t] = __builtin_amdgcn_wmma_f32_16x16x32_bf16(
          false, av, false, bv,
          (short)0, acc[t], /*reuse_a=*/false, /*reuse_b=*/false);
    }
  }

  // C layout: VGPR r, lanes 0-15 -> (M=r, N=lane); lanes 16-31 -> (M=8+r).
#pragma unroll
  for (int t = 0; t < 8; ++t) {
#pragma unroll
    for (int r = 0; r < 8; ++r) {
      const int row = m0 + r + hi * 8;
      const int col = n0 + t * 16 + lo;
      if (row < M) Wh[(size_t)row * N + col] = acc[t][r];
    }
  }
}

// ---------------------------------------------------------------------------
// Kernel 2: per-(node,head) attention logits s_i, s_j (32-wide dots).
// ---------------------------------------------------------------------------
__global__ void gat_node_scores(const float* __restrict__ Wh,
                                const float* __restrict__ attn,
                                float* __restrict__ s_i,
                                float* __restrict__ s_j, int nNH) {
  const int idx = blockIdx.x * blockDim.x + threadIdx.x;   // node*8 + h
  if (idx >= nNH) return;
  const int h = idx & 7;
  const float* __restrict__ whp = Wh + (size_t)idx * 32;
  const float* __restrict__ ai  = attn + h * 96;
  const float* __restrict__ aj  = ai + 32;
  float si = 0.f, sj = 0.f;
#pragma unroll
  for (int d = 0; d < 32; ++d) { const float v = whp[d]; si += v * ai[d]; sj += v * aj[d]; }
  s_i[idx] = si;
  s_j[idx] = sj;
}

// ---------------------------------------------------------------------------
// Kernel 3: fold We and be through a_e:  P[k,h] = sum_c We[k,c]*a_e[h,c],
// q[h] = sum_c be[c]*a_e[h,c].  One 256-thread block.
// ---------------------------------------------------------------------------
__global__ void gat_prep_edge_proj(const float* __restrict__ We,
                                   const float* __restrict__ attn,
                                   const float* __restrict__ be,
                                   float* __restrict__ P,
                                   float* __restrict__ q) {
  const int t = threadIdx.x;
  const int k = t >> 3, h = t & 7;
  const float* __restrict__ ae = attn + h * 96 + 64;
  float acc = 0.f;
#pragma unroll
  for (int c = 0; c < 32; ++c) acc += We[k * 32 + c] * ae[c];
  P[k * 8 + h] = acc;
  if (k == 0) {
    float qq = 0.f;
#pragma unroll
    for (int c = 0; c < 32; ++c) qq += be[c] * ae[c];
    q[h] = qq;
  }
}

// ---------------------------------------------------------------------------
// Kernel 4: zero the per-node max/sum tables.
// ---------------------------------------------------------------------------
__global__ void gat_zero_tables(float* __restrict__ nmax,
                                float* __restrict__ nsum, int nNH) {
  const int i = blockIdx.x * blockDim.x + threadIdx.x;
  if (i < nNH) { nmax[i] = 0.0f; nsum[i] = 0.0f; }
}

// ---------------------------------------------------------------------------
// Kernel 5: edge pass 1 — s_e + gathers + leaky_relu, store e_val,
// segment-max into nmax via int-ordered atomicMax (valid: nmax >= 0).
// ---------------------------------------------------------------------------
__global__ void gat_edge_pass1(const float* __restrict__ edge_attr,
                               const int* __restrict__ ei,
                               const float* __restrict__ s_i,
                               const float* __restrict__ s_j,
                               const float* __restrict__ P,
                               const float* __restrict__ q,
                               float* __restrict__ e_val,
                               float* __restrict__ nmax, int nE) {
  __shared__ float Psh[256];
  __shared__ float qsh[8];
  Psh[threadIdx.x] = P[threadIdx.x];
  if (threadIdx.x < 8) qsh[threadIdx.x] = q[threadIdx.x];
  __syncthreads();

  const int e = blockIdx.x * blockDim.x + threadIdx.x;
  if (e >= nE) return;
  const int src = ei[e];
  const int dst = ei[nE + e];

  float acc[8];
#pragma unroll
  for (int h = 0; h < 8; ++h) acc[h] = qsh[h];

  const float* __restrict__ ea = edge_attr + (size_t)e * 32;
#pragma unroll
  for (int c = 0; c < 32; ++c) {
    const float v = ea[c];
#pragma unroll
    for (int h = 0; h < 8; ++h) acc[h] += v * Psh[c * 8 + h];
  }

#pragma unroll
  for (int h = 0; h < 8; ++h) {
    float v = acc[h] + s_i[src * 8 + h] + s_j[dst * 8 + h];
    v = (v > 0.f) ? v : 0.2f * v;                     // leaky_relu(0.2)
    e_val[(size_t)e * 8 + h] = v;
    if (v > 0.0f)                                     // table clamped at 0
      atomicMax((int*)(nmax + (size_t)dst * 8 + h), __float_as_int(v));
  }
}

// ---------------------------------------------------------------------------
// Kernel 6: edge pass 2 — exp(e - max[dst]) and segment-sum into nsum.
// ---------------------------------------------------------------------------
__global__ void gat_edge_pass2(const int* __restrict__ ei,
                               const float* __restrict__ e_val,
                               const float* __restrict__ nmax,
                               float* __restrict__ nsum, int nE) {
  const int e = blockIdx.x * blockDim.x + threadIdx.x;
  if (e >= nE) return;
  const int dst = ei[nE + e];
#pragma unroll
  for (int h = 0; h < 8; ++h) {
    const float v  = e_val[(size_t)e * 8 + h];
    const float m  = nmax[(size_t)dst * 8 + h];
    const float ex = __expf(v - m);
    atomicAdd(nsum + (size_t)dst * 8 + h, ex);
  }
}

// ---------------------------------------------------------------------------
// Kernel 7: out[n,h,:] = Wh[n,h,:] * (S / max(S, 1e-10)), float4-wide.
// (Reference aggregates Wh[dst], so the weighted sum collapses to a scalar.)
// ---------------------------------------------------------------------------
__global__ void gat_finalize(const float4* __restrict__ Wh4,
                             const float* __restrict__ nsum,
                             float4* __restrict__ out4, int total4) {
  const int idx = blockIdx.x * blockDim.x + threadIdx.x;
  if (idx >= total4) return;
  const int nh = idx >> 3;                    // 8 float4s per (node,head)
  const float S = nsum[nh];
  const float f = S / fmaxf(S, 1e-10f);
  float4 w = Wh4[idx];
  w.x *= f; w.y *= f; w.z *= f; w.w *= f;
  out4[idx] = w;
}

// ---------------------------------------------------------------------------
extern "C" void kernel_launch(void* const* d_in, const int* in_sizes, int n_in,
                              void* d_out, int out_size, void* d_ws, size_t ws_size,
                              hipStream_t stream) {
  (void)n_in; (void)ws_size; (void)out_size;
  const float* x         = (const float*)d_in[0];
  const int*   edge_idx  = (const int*)  d_in[1];
  const float* edge_attr = (const float*)d_in[2];
  const float* W         = (const float*)d_in[3];
  const float* attn      = (const float*)d_in[4];
  const float* We        = (const float*)d_in[5];
  const float* be        = (const float*)d_in[6];
  float*       out       = (float*)d_out;

  const int IN_DIM = 512, HD = 256;
  const int nNodes = in_sizes[0] / IN_DIM;     // 50000
  const int nE     = in_sizes[1] / 2;          // 800000
  const int nNH    = nNodes * 8;

  // Workspace carve-up (256B aligned).
  char* ws = (char*)d_ws;
  size_t off = 0;
  auto carve = [&](size_t bytes) -> char* {
    char* p = ws + off;
    off += (bytes + 255) & ~(size_t)255;
    return p;
  };
  float*  Wh    = (float*) carve((size_t)nNodes * HD * sizeof(float));
  __bf16* Wb    = (__bf16*)carve((size_t)IN_DIM * HD * sizeof(__bf16));
  float*  s_i   = (float*) carve((size_t)nNH * sizeof(float));
  float*  s_j   = (float*) carve((size_t)nNH * sizeof(float));
  float*  P     = (float*) carve(256 * sizeof(float));
  float*  q     = (float*) carve(8 * sizeof(float));
  float*  nmax  = (float*) carve((size_t)nNH * sizeof(float));
  float*  nsum  = (float*) carve((size_t)nNH * sizeof(float));
  float*  e_val = (float*) carve((size_t)nE * 8 * sizeof(float));

  // 0) W -> bf16 (one-time, 256 KB; L2-resident for the GEMM)
  {
    const int total = IN_DIM * HD;
    gat_cvt_w_bf16<<<(total + 255) / 256, 256, 0, stream>>>(W, Wb, total);
  }
  // 1) WMMA GEMM: Wh = x @ W   (64x256 block tile, 8 WMMAs per A fragment)
  gat_gemm_wmma_bf16<<<(nNodes + 63) / 64, 256, 0, stream>>>(x, Wb, Wh, nNodes);
  // 2) s_i, s_j
  gat_node_scores<<<(nNH + 255) / 256, 256, 0, stream>>>(Wh, attn, s_i, s_j, nNH);
  // 3) P, q
  gat_prep_edge_proj<<<1, 256, 0, stream>>>(We, attn, be, P, q);
  // 4) zero tables
  gat_zero_tables<<<(nNH + 255) / 256, 256, 0, stream>>>(nmax, nsum, nNH);
  // 5) edge pass 1
  gat_edge_pass1<<<(nE + 255) / 256, 256, 0, stream>>>(edge_attr, edge_idx, s_i, s_j,
                                                       P, q, e_val, nmax, nE);
  // 6) edge pass 2
  gat_edge_pass2<<<(nE + 255) / 256, 256, 0, stream>>>(edge_idx, e_val, nmax, nsum, nE);
  // 7) finalize
  {
    const int total4 = nNodes * (HD / 4);
    gat_finalize<<<(total4 + 255) / 256, 256, 0, stream>>>((const float4*)Wh, nsum,
                                                           (float4*)out, total4);
  }
}
